// GNN_55576876810911
// MI455X (gfx1250) — matrix operand order repaired and checked by reading, not compiled, not run on hardware
//
#include <hip/hip_runtime.h>

#define N_NODES 50000
#define N_EDGES 800000
#define IN_DIM  5
#define HIDDEN  128
#define OUT_DIM 60
#define OUT_PAD 64
#define LDSP    67   // padded LDS row stride in dwords (128 bf16 = 64 dwords + 3)

typedef __attribute__((ext_vector_type(16))) __bf16 v16bf;
typedef __attribute__((ext_vector_type(8)))  float  v8f;

union FragBF { v16bf v; unsigned int u[8]; };

__device__ __forceinline__ unsigned short f2bf(float f) {
  unsigned int u = __float_as_uint(f);
  u += 0x7FFFu + ((u >> 16) & 1u);          // round-to-nearest-even
  return (unsigned short)(u >> 16);
}
__device__ __forceinline__ float bf2f(unsigned short h) {
  return __uint_as_float(((unsigned int)h) << 16);
}

// Build A fragment (16x32 bf16) from an LDS tile with row stride LDSP dwords.
// ISA 7.12.2: lane m=lane%16; VGPR v0..3 hold K=(lane/16)*8+{0..7},
// VGPR v4..7 hold K=16+(lane/16)*8+{0..7} (bf16 pairs per dword).
__device__ __forceinline__ FragBF frag_a_lds(const unsigned int* lds, int k0, int lane) {
  FragBF f;
  const unsigned int* b = lds + (lane & 15) * LDSP + (k0 >> 1) + (lane >> 4) * 4;
#pragma unroll
  for (int v = 0; v < 4; ++v) {
    f.u[v]     = b[v];       // K = k0 + q*8 + 2v   (dword = k0/2 + q*4 + v)
    f.u[v + 4] = b[v + 8];   // +16 bf16 = +8 dwords
  }
  return f;
}

// B fragment (32x16 bf16, K x N) where B[k,n] = W[n,k], W row-major [N,K] ld=K.
// Lane n=lane%16; VGPR v0..3 hold K=(lane/16)*8+{0..7}, v4..7 hold K=16+(lane/16)*8+{0..7} as bf16 pairs.
__device__ __forceinline__ FragBF load_b_bf(const unsigned short* W, int n0, int k0,
                                            int ld, int lane) {
  FragBF f;
  const unsigned short* row = W + (n0 + (lane & 15)) * ld;
  int h = (lane >> 4) * 8;
#pragma unroll
  for (int v = 0; v < 8; ++v)
    f.u[v] = *(const unsigned int*)(row + k0 + h + 2 * v + (v >= 4 ? 8 : 0));
  return f;
}

__device__ __forceinline__ v8f wmma_bf16(FragBF a, FragBF b, v8f c) {
  return __builtin_amdgcn_wmma_f32_16x16x32_bf16(false, a.v, false, b.v,
                                                 (short)0, c, false, false);
}

// Stage one 16x128 bf16 tile (1024 dwords) into LDS, `per` dwords per thread.
__device__ __forceinline__ void stage_tile(const unsigned short* src, int m0,
                                           unsigned int* dst, int tid, int per) {
  int idx = tid * per;                       // per is 4 or 8; covers 1024 dwords
#pragma unroll
  for (int j = 0; j < 2; ++j) {              // up to two uint4 chunks
    if (j * 4 >= per) break;
    int i = idx + j * 4;
    int r = i >> 6, c = i & 63;
    uint4 v = *(const uint4*)((const unsigned int*)(src + (m0 + r) * HIDDEN) + c);
    unsigned int* d = dst + r * LDSP + c;
    d[0] = v.x; d[1] = v.y; d[2] = v.z; d[3] = v.w;
  }
}

// ---------------------------------------------------------------- utility
__global__ void fill_zero(float* p, int n) {
  int i = blockIdx.x * blockDim.x + threadIdx.x;
  if (i < n) p[i] = 0.0f;
}

__global__ void cvt_f32_bf16(const float* src, unsigned int* dst, int npairs) {
  int i = blockIdx.x * blockDim.x + threadIdx.x;
  if (i < npairs) {
    float2 p = *(const float2*)(src + 2 * i);
    dst[i] = (unsigned)f2bf(p.x) | ((unsigned)f2bf(p.y) << 16);
  }
}

__global__ void cvt_weights(const float* w2s, const float* w2n,
                            const float* w3s, const float* w3n,
                            unsigned short* w2s_bf, unsigned short* w2n_bf,
                            unsigned short* w3s_bf, unsigned short* w3n_bf) {
  int i = blockIdx.x * blockDim.x + threadIdx.x;
  if (i < HIDDEN * HIDDEN) {
    w2s_bf[i] = f2bf(w2s[i]);
    w2n_bf[i] = f2bf(w2n[i]);
  }
  if (i < OUT_PAD * HIDDEN) {
    int n = i / HIDDEN;
    w3s_bf[i] = f2bf(n < OUT_DIM ? w3s[i] : 0.0f);  // pad rows 60..63 with 0
    w3n_bf[i] = f2bf(n < OUT_DIM ? w3n[i] : 0.0f);
  }
}

// ---------------------------------------------------------------- layer 1
__global__ void agg1_k(const int* ei, const float* x, float* agg1) {
  int e = blockIdx.x * blockDim.x + threadIdx.x;
  if (e >= N_EDGES) return;
  int s = ei[e], d = ei[N_EDGES + e];
#pragma unroll
  for (int k = 0; k < IN_DIM; ++k)
    atomicAdd(agg1 + d * IN_DIM + k, x[s * IN_DIM + k]);
}

__global__ void layer1_k(const float* x, const float* agg1,
                         const float* w1s, const float* w1n, const float* b1,
                         unsigned short* h1bf) {
  int n = blockIdx.x;
  int f = threadIdx.x;
  float acc = b1[f];
#pragma unroll
  for (int k = 0; k < IN_DIM; ++k)
    acc += x[n * IN_DIM + k] * w1s[f * IN_DIM + k] +
           agg1[n * IN_DIM + k] * w1n[f * IN_DIM + k];
  h1bf[n * HIDDEN + f] = f2bf(fmaxf(acc, 0.0f));
}

// ------------------------------------------------- 128-dim aggregation (bf16 gather)
__global__ void agg_bf128(const int* ei, const unsigned short* h, float* agg) {
  int t = blockIdx.x * blockDim.x + threadIdx.x;
  int e = t >> 5, lane = t & 31;
  if (e >= N_EDGES) return;
  int s = ei[e], d = ei[N_EDGES + e];
  const unsigned int* p = (const unsigned int*)(h + s * HIDDEN) + lane * 2;
  unsigned int u0 = p[0], u1 = p[1];              // 4 bf16 values
  float* q = agg + d * HIDDEN + lane * 4;
  atomicAdd(q + 0, bf2f((unsigned short)(u0 & 0xFFFF)));
  atomicAdd(q + 1, bf2f((unsigned short)(u0 >> 16)));
  atomicAdd(q + 2, bf2f((unsigned short)(u1 & 0xFFFF)));
  atomicAdd(q + 3, bf2f((unsigned short)(u1 >> 16)));
}

// ------------------------------------------------- layer 2 GEMM (WMMA bf16, LDS-staged A)
__global__ void gemm2_k(const unsigned short* h1bf, const unsigned short* agg2bf,
                        const unsigned short* w2s_bf, const unsigned short* w2n_bf,
                        const float* b2, unsigned short* h2bf) {
  __shared__ unsigned int lds[2 * 16 * LDSP];
  int tid  = threadIdx.x;
  int lane = tid & 31;
  int wave = tid >> 5;             // 8 waves -> 8 N-tiles (N=128)
  int m0 = blockIdx.x * 16;
  int n0 = wave * 16;

  stage_tile(h1bf,   m0, lds,             tid, 4);   // 256 threads x 4 dwords
  stage_tile(agg2bf, m0, lds + 16 * LDSP, tid, 4);
  __syncthreads();

  v8f c = {};
#pragma unroll
  for (int k0 = 0; k0 < HIDDEN; k0 += 32) {
    FragBF as = frag_a_lds(lds, k0, lane);
    FragBF bs = load_b_bf(w2s_bf, n0, k0, HIDDEN, lane);
    c = wmma_bf16(as, bs, c);
    FragBF an = frag_a_lds(lds + 16 * LDSP, k0, lane);
    FragBF bn = load_b_bf(w2n_bf, n0, k0, HIDDEN, lane);
    c = wmma_bf16(an, bn, c);
  }
  int n = n0 + (lane & 15);
  int half = lane >> 4;
  float bias = b2[n];
#pragma unroll
  for (int v = 0; v < 8; ++v) {
    int m = m0 + v + 8 * half;
    h2bf[m * HIDDEN + n] = f2bf(fmaxf(c[v] + bias, 0.0f));
  }
}

// ------------------------------------------------- t3 = h2 @ w3_nbr^T (WMMA, LDS-staged A)
__global__ void gemm3_k(const unsigned short* h2bf, const unsigned short* w3n_bf,
                        float* t3) {
  __shared__ unsigned int lds[16 * LDSP];
  int tid  = threadIdx.x;
  int lane = tid & 31;
  int wave = tid >> 5;             // 4 waves -> 4 N-tiles (N=64)
  int m0 = blockIdx.x * 16;
  int n0 = wave * 16;

  stage_tile(h2bf, m0, lds, tid, 8);                 // 128 threads x 8 dwords
  __syncthreads();

  v8f c = {};
#pragma unroll
  for (int k0 = 0; k0 < HIDDEN; k0 += 32) {
    FragBF a = frag_a_lds(lds, k0, lane);
    FragBF b = load_b_bf(w3n_bf, n0, k0, HIDDEN, lane);
    c = wmma_bf16(a, b, c);
  }
  int n = n0 + (lane & 15);
  int half = lane >> 4;
#pragma unroll
  for (int v = 0; v < 8; ++v)
    t3[(m0 + v + 8 * half) * OUT_PAD + n] = c[v];
}

// ------------------------------------------------- 64-dim aggregation of t3
__global__ void agg3_k(const int* ei, const float* t3, float* agg3) {
  int t = blockIdx.x * blockDim.x + threadIdx.x;
  int e = t >> 4, l = t & 15;
  if (e >= N_EDGES) return;
  int s = ei[e], d = ei[N_EDGES + e];
  float4 v = *(const float4*)(t3 + s * OUT_PAD + l * 4);
  float* q = agg3 + d * OUT_PAD + l * 4;
  atomicAdd(q + 0, v.x);
  atomicAdd(q + 1, v.y);
  atomicAdd(q + 2, v.z);
  atomicAdd(q + 3, v.w);
}

// ------------------------------------------------- out = h2@w3_self^T + agg3 + b3
__global__ void final_k(const unsigned short* h2bf, const unsigned short* w3s_bf,
                        const float* agg3, const float* b3, float* out) {
  __shared__ unsigned int lds[16 * LDSP];
  int tid  = threadIdx.x;
  int lane = tid & 31;
  int wave = tid >> 5;             // 4 waves -> 4 N-tiles (N=64 padded)
  int m0 = blockIdx.x * 16;
  int n0 = wave * 16;

  stage_tile(h2bf, m0, lds, tid, 8);
  __syncthreads();

  int n = n0 + (lane & 15);
  int half = lane >> 4;
  float bias = (n < OUT_DIM) ? b3[n] : 0.0f;
  v8f c;
#pragma unroll
  for (int v = 0; v < 8; ++v)
    c[v] = agg3[(m0 + v + 8 * half) * OUT_PAD + n] + bias;
#pragma unroll
  for (int k0 = 0; k0 < HIDDEN; k0 += 32) {
    FragBF a = frag_a_lds(lds, k0, lane);
    FragBF b = load_b_bf(w3s_bf, n0, k0, HIDDEN, lane);
    c = wmma_bf16(a, b, c);
  }
#pragma unroll
  for (int v = 0; v < 8; ++v) {
    int m = m0 + v + 8 * half;
    if (n < OUT_DIM) out[m * OUT_DIM + n] = c[v];
  }
}

extern "C" void kernel_launch(void* const* d_in, const int* in_sizes, int n_in,
                              void* d_out, int out_size, void* d_ws, size_t ws_size,
                              hipStream_t stream) {
  const float* x   = (const float*)d_in[0];
  const int*   ei  = (const int*)d_in[1];
  const float* w1s = (const float*)d_in[2];
  const float* w1n = (const float*)d_in[3];
  const float* b1  = (const float*)d_in[4];
  const float* w2s = (const float*)d_in[5];
  const float* w2n = (const float*)d_in[6];
  const float* b2  = (const float*)d_in[7];
  const float* w3s = (const float*)d_in[8];
  const float* w3n = (const float*)d_in[9];
  const float* b3  = (const float*)d_in[10];
  float* out = (float*)d_out;

  char* p = (char*)d_ws;
  auto carve = [&](size_t bytes) -> void* {
    void* r = (void*)p;
    p += (bytes + 255) & ~(size_t)255;
    return r;
  };
  unsigned short* h1bf   = (unsigned short*)carve((size_t)N_NODES * HIDDEN * 2);
  float*          agg2   = (float*)carve((size_t)N_NODES * HIDDEN * 4);
  unsigned short* agg2bf = (unsigned short*)carve((size_t)N_NODES * HIDDEN * 2);
  unsigned short* h2bf   = (unsigned short*)carve((size_t)N_NODES * HIDDEN * 2);
  float*          agg1   = (float*)carve((size_t)N_NODES * IN_DIM * 4);
  unsigned short* w2s_bf = (unsigned short*)carve(HIDDEN * HIDDEN * 2);
  unsigned short* w2n_bf = (unsigned short*)carve(HIDDEN * HIDDEN * 2);
  unsigned short* w3s_bf = (unsigned short*)carve(OUT_PAD * HIDDEN * 2);
  unsigned short* w3n_bf = (unsigned short*)carve(OUT_PAD * HIDDEN * 2);
  float* t3   = (float*)h1bf;   // alias: h1bf dead after gemm2_k
  float* agg3 = agg2;           // alias: agg2 (f32) dead after cvt_f32_bf16

  fill_zero<<<(N_NODES * IN_DIM + 255) / 256, 256, 0, stream>>>(agg1, N_NODES * IN_DIM);
  fill_zero<<<(N_NODES * HIDDEN + 255) / 256, 256, 0, stream>>>(agg2, N_NODES * HIDDEN);
  cvt_weights<<<(HIDDEN * HIDDEN + 255) / 256, 256, 0, stream>>>(
      w2s, w2n, w3s, w3n, w2s_bf, w2n_bf, w3s_bf, w3n_bf);

  agg1_k<<<(N_EDGES + 255) / 256, 256, 0, stream>>>(ei, x, agg1);
  layer1_k<<<N_NODES, HIDDEN, 0, stream>>>(x, agg1, w1s, w1n, b1, h1bf);

  agg_bf128<<<(N_EDGES * 32) / 256, 256, 0, stream>>>(ei, h1bf, agg2);
  cvt_f32_bf16<<<(N_NODES * HIDDEN / 2 + 255) / 256, 256, 0, stream>>>(
      agg2, (unsigned int*)agg2bf, N_NODES * HIDDEN / 2);
  fill_zero<<<(N_NODES * OUT_PAD + 255) / 256, 256, 0, stream>>>(agg3, N_NODES * OUT_PAD);

  gemm2_k<<<N_NODES / 16, 256, 0, stream>>>(h1bf, agg2bf, w2s_bf, w2n_bf, b2, h2bf);
  gemm3_k<<<N_NODES / 16, 128, 0, stream>>>(h2bf, w3n_bf, t3);
  agg3_k<<<(N_EDGES * 16) / 256, 256, 0, stream>>>(ei, t3, agg3);
  final_k<<<N_NODES / 16, 128, 0, stream>>>(h2bf, w3s_bf, agg3, b3, out);
}
